// TTTBase_42253888258296
// MI455X (gfx1250) — compile-verified, matrix operand order checked by reading, not run on hardware
//
#include <hip/hip_runtime.h>
#include <cstdint>

#define B_   4
#define L_   2048
#define W_   2048
#define NH_  32
#define HD_  64
#define MBS_ 16
#define NMB_ 128
#define EPS_ 1e-6f

typedef __attribute__((ext_vector_type(2)))  float        v2f;
typedef __attribute__((ext_vector_type(8)))  float        v8f;
typedef __attribute__((ext_vector_type(16))) __bf16       v16bf;
typedef __attribute__((ext_vector_type(4)))  unsigned int u32x4;
typedef __attribute__((ext_vector_type(4)))  int          i32x4;
typedef __attribute__((ext_vector_type(8)))  int          i32x8;

#if __has_builtin(__builtin_amdgcn_tensor_load_to_lds)
#define HAVE_TDM 1
#else
#define HAVE_TDM 0
#endif

#if __has_builtin(__builtin_amdgcn_s_wait_tensorcnt)
#define WAIT_TENSORCNT(n) __builtin_amdgcn_s_wait_tensorcnt(n)
#else
#define WAIT_TENSORCNT(n) asm volatile("s_wait_tensorcnt %0" :: "i"(n) : "memory")
#endif

// ---- WMMA wrappers (CDNA5 gfx1250) ----
static __device__ inline v8f wmma_f32k4(v2f a, v2f b, v8f c) {
  return __builtin_amdgcn_wmma_f32_16x16x4_f32(false, a, false, b, (short)0, c, false, false);
}
static __device__ inline v8f wmma_bf16(v16bf a, v16bf b, v8f c) {
  return __builtin_amdgcn_wmma_f32_16x16x32_bf16(false, a, false, b, (short)0, c, false, false);
}

// 16-bit operand packing per CDNA5 ISA (A 16x32): element j of v16bf for lane L
static __device__ inline int kmap16(int lane, int j) {
  int r = j >> 1, h2 = j & 1;
  return ((r >> 2) << 4) + ((lane >> 4) << 3) + ((r & 3) << 1) + h2;
}

// =====================================================================
// TDM: async-load a 128x32 fp32 tile (row stride K floats) into LDS with
// 1-dword padding every 32 dwords (LDS row stride 33). D# per ISA 08 §8.
// =====================================================================
#define STG_STRIDE 33
#define STG_TILE   (128 * STG_STRIDE)    // floats per staged tile

static __device__ inline void tdm_load_tile(const float* gsrc, unsigned lds_byte_off, int K) {
#if HAVE_TDM
  unsigned long long ga = (unsigned long long)(uintptr_t)gsrc;
  u32x4 g0;
  g0[0] = 1u;                                            // count=1 (valid user D#)
  g0[1] = lds_byte_off;                                  // lds_addr
  g0[2] = (unsigned)(ga & 0xFFFFFFFFu);                  // global_addr[31:0]
  g0[3] = (unsigned)((ga >> 32) & 0x01FFFFFFu) | (2u << 30); // global_addr[56:32], type=2
  i32x8 g1;
  g1[0] = (int)((2u << 16) | (1u << 20) | (4u << 22));   // data_size=4B, pad_en, interval=32dw (+1dw)
  g1[1] = (int)((unsigned)K << 16);                      // tensor_dim0 = K (low16)
  g1[2] = (int)(0xFFFFu << 16) | (int)((unsigned)K >> 16); // tensor_dim1 generous, dim0 hi
  g1[3] = (int)(32u << 16);                              // tile_dim0 = 32 (dim1 hi = 0)
  g1[4] = 128;                                           // tile_dim1 = 128, tile_dim2 = 0
  g1[5] = K;                                             // tensor_dim0_stride[31:0]
  g1[6] = 0;                                             // stride0 hi, stride1 lo
  g1[7] = 0;
  i32x4 z4; z4[0]=0; z4[1]=0; z4[2]=0; z4[3]=0;
#if defined(__clang_major__) && __clang_major__ >= 23
  i32x8 z8; for (int i = 0; i < 8; ++i) z8[i] = 0;
  __builtin_amdgcn_tensor_load_to_lds(g0, g1, z4, z4, z8, 0);
#else
  __builtin_amdgcn_tensor_load_to_lds(g0, g1, z4, z4, 0);
#endif
#else
  (void)gsrc; (void)lds_byte_off; (void)K;
#endif
}

// =====================================================================
// GEMM: C[M,N] = A[M,K] @ W[N,K]^T with split-bf16 (3x bf16 WMMA ~ fp32),
// TDM double-buffered staging. storeMode 0: row-major; 1: (B,NH,NMB,MBS,HD)
// Dynamic LDS: 4 * STG_TILE floats (bufA0,bufB0,bufA1,bufB1) = 67.6 KB
// =====================================================================
__global__ __launch_bounds__(256) void gemm_bf16split(
    const float* __restrict__ A, const float* __restrict__ Wm,
    float* __restrict__ C, int M, int N, int K, int storeMode)
{
  extern __shared__ float dynls[];
  const int tid  = threadIdx.x;
  const int lane = tid & 31;
  const int wv   = tid >> 5;          // 8 waves
  const int wm   = wv >> 1;           // 0..3  (32-row stripe)
  const int wn   = wv & 1;            // 0..1  (64-col stripe)
  const int m0   = blockIdx.y * 128;
  const int n0   = blockIdx.x * 128;
  const unsigned ldsbase = __builtin_amdgcn_groupstaticsize();

  v8f acc[2][4];
  for (int i = 0; i < 2; ++i)
    for (int j = 0; j < 4; ++j)
      for (int r = 0; r < 8; ++r) acc[i][j][r] = 0.f;

  const int nk = K >> 5;

#if HAVE_TDM
  if (wv == 0) {   // wave-uniform: TDM ignores EXEC, one issue per wave
    tdm_load_tile(A  + (size_t)m0 * K, ldsbase + 0u,                           K);
    tdm_load_tile(Wm + (size_t)n0 * K, ldsbase + (unsigned)(STG_TILE * 4),     K);
  }
#endif

  for (int ck = 0; ck < nk; ++ck) {
    const int buf = ck & 1;
    float* stA = dynls + buf * (2 * STG_TILE);
    float* stB = stA + STG_TILE;

#if HAVE_TDM
    if (wv == 0) {
      if (ck + 1 < nk) {
        int nb = (ck + 1) & 1;
        tdm_load_tile(A  + (size_t)m0 * K + (ck + 1) * 32,
                      ldsbase + (unsigned)((nb * 2 + 0) * STG_TILE * 4), K);
        tdm_load_tile(Wm + (size_t)n0 * K + (ck + 1) * 32,
                      ldsbase + (unsigned)((nb * 2 + 1) * STG_TILE * 4), K);
        WAIT_TENSORCNT(2);   // chunk ck's 2 loads done (in-order), ck+1 in flight
      } else {
        WAIT_TENSORCNT(0);
      }
    }
    __syncthreads();
#else
    // fallback: cooperative staged load (no pipelining)
    __syncthreads();
    for (int q = 0; q < 4; ++q) {
      int s   = tid + 256 * q;           // 1024 float4 slots
      int row = s >> 3;
      int c4  = (s & 7) << 2;
      float4 av = *(const float4*)(A  + (size_t)(m0 + row) * K + ck * 32 + c4);
      float4 bv = *(const float4*)(Wm + (size_t)(n0 + row) * K + ck * 32 + c4);
      stA[row * STG_STRIDE + c4 + 0] = av.x; stA[row * STG_STRIDE + c4 + 1] = av.y;
      stA[row * STG_STRIDE + c4 + 2] = av.z; stA[row * STG_STRIDE + c4 + 3] = av.w;
      stB[row * STG_STRIDE + c4 + 0] = bv.x; stB[row * STG_STRIDE + c4 + 1] = bv.y;
      stB[row * STG_STRIDE + c4 + 2] = bv.z; stB[row * STG_STRIDE + c4 + 3] = bv.w;
    }
    __syncthreads();
#endif

    const int mr = lane & 15;
    v16bf ahf[2], alf[2];
    for (int mt = 0; mt < 2; ++mt) {
      const float* rp = stA + (size_t)(wm * 32 + mt * 16 + mr) * STG_STRIDE;
      for (int j = 0; j < 16; ++j) {
        float f = rp[kmap16(lane, j)];
        __bf16 hi = (__bf16)f;
        ahf[mt][j] = hi;
        alf[mt][j] = (__bf16)(f - (float)hi);
      }
    }
    for (int nt = 0; nt < 4; ++nt) {
      const float* cp = stB + (size_t)(wn * 64 + nt * 16 + mr) * STG_STRIDE;
      v16bf bh, bl;
      for (int j = 0; j < 16; ++j) {
        float f = cp[kmap16(lane, j)];
        __bf16 hi = (__bf16)f;
        bh[j] = hi;
        bl[j] = (__bf16)(f - (float)hi);
      }
      for (int mt = 0; mt < 2; ++mt) {
        acc[mt][nt] = wmma_bf16(ahf[mt], bh, acc[mt][nt]);
        acc[mt][nt] = wmma_bf16(ahf[mt], bl, acc[mt][nt]);
        acc[mt][nt] = wmma_bf16(alf[mt], bh, acc[mt][nt]);
      }
    }
    __syncthreads();   // all waves done with buf before TDM rewrites it
  }

  // store (C layout: n = lane&15, m = r + (lane>=16 ? 8:0))
  const int cn = lane & 15;
  const int ch = lane >> 4;
  for (int mt = 0; mt < 2; ++mt) {
    for (int nt = 0; nt < 4; ++nt) {
      int gn = n0 + wn * 64 + nt * 16 + cn;
      for (int r = 0; r < 8; ++r) {
        int gm = m0 + wm * 32 + mt * 16 + r + (ch << 3);
        float v = acc[mt][nt][r];
        if (storeMode == 0) {
          C[(size_t)gm * N + gn] = v;
        } else {
          int b = gm >> 11, l = gm & (L_ - 1);
          int h = gn >> 6,  d = gn & 63;
          size_t idx = ((((size_t)b * NH_ + h) * NMB_ + (l >> 4)) * MBS_ + (l & 15)) * HD_ + d;
          C[idx] = v;
        }
      }
    }
  }
}

// =====================================================================
// Causal depthwise conv (CK=4) + interleaved RoPE -> head/chunk layout
// =====================================================================
__global__ __launch_bounds__(256) void conv_rope(
    const float* __restrict__ xq,
    const float* __restrict__ cqw, const float* __restrict__ cqb,
    const float* __restrict__ ckw, const float* __restrict__ ckb,
    float* __restrict__ XQh, float* __restrict__ XKh)
{
  int pid = blockIdx.x * 256 + threadIdx.x;  // B*L*W/2 threads
  int d2 = pid & 31;
  int h  = (pid >> 5) & 31;
  int l  = (pid >> 10) & (L_ - 1);
  int b  = pid >> 21;
  int w0 = h * 64 + d2 * 2;

  float qe = cqb[w0], qo = cqb[w0 + 1], ke = ckb[w0], ko = ckb[w0 + 1];
  for (int k = 0; k < 4; ++k) {
    int ll = l + k - 3;
    if (ll < 0) continue;
    size_t si = ((size_t)(b * L_ + ll)) * W_ + w0;
    float xe = xq[si], xo = xq[si + 1];
    qe += xe * cqw[w0 * 4 + k];       qo += xo * cqw[(w0 + 1) * 4 + k];
    ke += xe * ckw[w0 * 4 + k];       ko += xo * ckw[(w0 + 1) * 4 + k];
  }
  float ang = (float)(l & 15) * __expf(-(float)d2 * 0.2878231366242558f); // ln(1e4)/32
  float sn, cs;
  __sincosf(ang, &sn, &cs);
  float Qe = qe * cs - qo * sn, Qo = qo * cs + qe * sn;
  float Ke = ke * cs - ko * sn, Ko = ko * cs + ke * sn;
  size_t oidx = ((((size_t)b * NH_ + h) * NMB_ + (l >> 4)) * MBS_ + (l & 15)) * HD_ + 2 * d2;
  XQh[oidx] = Qe; XQh[oidx + 1] = Qo;
  XKh[oidx] = Ke; XKh[oidx + 1] = Ko;
}

// =====================================================================
// ttt_lr[b,h,l] = sigmoid(<hs[b,l,:], ttt_lr_w[h,0,:]> + ttt_lr_b[h])
// =====================================================================
__global__ __launch_bounds__(256) void ttt_lr_k(
    const float* __restrict__ hs, const float* __restrict__ lrw,
    const float* __restrict__ lrb, float* __restrict__ outp)
{
  __shared__ float rowb[2048];
  __shared__ float red[32][9];
  int blid = blockIdx.x;   // b*L + l
  int tid  = threadIdx.x;
  size_t base = (size_t)blid * W_;
  const float4* src = (const float4*)(hs + base);
  float4* dst = (float4*)rowb;
  dst[tid] = src[tid];
  dst[tid + 256] = src[tid + 256];
  __syncthreads();
  int h = tid >> 3, sg = tid & 7;
  const float* wrow = lrw + (size_t)h * W_;
  float s = 0.f;
  for (int c = sg; c < W_; c += 8) s += rowb[c] * wrow[c];
  red[h][sg] = s;
  __syncthreads();
  if (tid < 32) {
    float t = 0.f;
    for (int q = 0; q < 8; ++q) t += red[tid][q];
    t += lrb[tid];
    int b = blid >> 11, l = blid & (L_ - 1);
    outp[((size_t)(b * NH_ + tid)) * L_ + l] = 1.0f / (1.0f + __expf(-t));
  }
}

// =====================================================================
// TTT scan: one block per (b,h); 4 waves; sequential over 128 minibatches.
// All matmuls via v_wmma_f32_16x16x4_f32; LN reductions on VALU.
// =====================================================================
__global__ __launch_bounds__(128) void ttt_scan(
    const float* __restrict__ XQ, const float* __restrict__ XK, const float* __restrict__ XV,
    const float* __restrict__ lrbuf, const float* __restrict__ lti,
    const float* __restrict__ nw, const float* __restrict__ nb,
    const float* __restrict__ W1in, const float* __restrict__ b1in,
    float* __restrict__ outp)
{
  __shared__ float W1s[64][65];
  __shared__ float xqs[16][68];
  __shared__ float xks[16][68];
  __shared__ float xvs[16][68];
  __shared__ float Z1s[16][68];
  __shared__ float grads[16][68];
  __shared__ float Ms[16][16];
  __shared__ float b1s[64];
  __shared__ float gW[64], gB[64];
  __shared__ float lrv[16], tokv[16], lev[16];
  __shared__ float red1[16][8], red2[16][8];

  const int tid  = threadIdx.x;
  const int bh   = blockIdx.x;          // b*NH + h
  const int b    = bh >> 5, h = bh & 31;
  const int lane = tid & 31, wv = tid >> 5;
  const int ln16 = lane & 15, kh = lane >> 4;

  for (int i = tid; i < 64 * 64; i += 128) W1s[i >> 6][i & 63] = W1in[(size_t)h * 4096 + i];
  if (tid < 64) { b1s[tid] = b1in[h * 64 + tid]; gW[tid] = nw[h * 64 + tid]; gB[tid] = nb[h * 64 + tid]; }
  if (tid < 16) tokv[tid] = fmaxf(1.0f / (float)(tid + 1) + lti[tid], 0.0f);
  __syncthreads();

  const size_t base0 = (size_t)bh * NMB_ * MBS_ * HD_;
  for (int n = 0; n < NMB_; ++n) {
    const size_t tb = base0 + (size_t)n * MBS_ * HD_;
    {
      int idx = tid * 8;
      int row = idx >> 6, col = idx & 63;
      float4 a0 = *(const float4*)(XQ + tb + idx);
      float4 a1 = *(const float4*)(XQ + tb + idx + 4);
      float4 k0 = *(const float4*)(XK + tb + idx);
      float4 k1 = *(const float4*)(XK + tb + idx + 4);
      float4 v0 = *(const float4*)(XV + tb + idx);
      float4 v1 = *(const float4*)(XV + tb + idx + 4);
      xqs[row][col+0]=a0.x; xqs[row][col+1]=a0.y; xqs[row][col+2]=a0.z; xqs[row][col+3]=a0.w;
      xqs[row][col+4]=a1.x; xqs[row][col+5]=a1.y; xqs[row][col+6]=a1.z; xqs[row][col+7]=a1.w;
      xks[row][col+0]=k0.x; xks[row][col+1]=k0.y; xks[row][col+2]=k0.z; xks[row][col+3]=k0.w;
      xks[row][col+4]=k1.x; xks[row][col+5]=k1.y; xks[row][col+6]=k1.z; xks[row][col+7]=k1.w;
      xvs[row][col+0]=v0.x; xvs[row][col+1]=v0.y; xvs[row][col+2]=v0.z; xvs[row][col+3]=v0.w;
      xvs[row][col+4]=v1.x; xvs[row][col+5]=v1.y; xvs[row][col+6]=v1.z; xvs[row][col+7]=v1.w;
    }
    if (tid < 16) {
      float lrval = lrbuf[(size_t)bh * L_ + n * 16 + tid];
      lrv[tid] = lrval;
      lev[tid] = tokv[15] * lrval * (1.0f / 64.0f);   // last_eta
    }
    __syncthreads();

    // ---- Z1 = xk @ W1 + b1 : wave wv owns cols [16wv,16wv+16) ----
    {
      v8f a8; for (int r = 0; r < 8; ++r) a8[r] = 0.f;
      int c = wv * 16 + ln16;
      for (int ks = 0; ks < 16; ++ks) {
        int k = ks * 4 + kh * 2;
        v2f av, bv;
        av.x = xks[ln16][k]; av.y = xks[ln16][k + 1];
        bv.x = W1s[k][c];    bv.y = W1s[k + 1][c];
        a8 = wmma_f32k4(av, bv, a8);
      }
      for (int r = 0; r < 8; ++r) { int m = r + (kh << 3); Z1s[m][c] = a8[r] + b1s[c]; }
    }
    __syncthreads();

    // ---- grad = ln_l2_bwd(Z1, xv - xk, g, b) ----
    {
      int row = tid >> 3, sg = tid & 7;
      float z[8], tg[8];
      float s1 = 0.f, s2 = 0.f;
      for (int q = 0; q < 8; ++q) {
        int c = sg * 8 + q;
        float v = Z1s[row][c];
        z[q] = v; s1 += v; s2 += v * v;
        tg[q] = xvs[row][c] - xks[row][c];
      }
      red1[row][sg] = s1; red2[row][sg] = s2;
      __syncthreads();
      float mu = 0.f, m2 = 0.f;
      for (int q = 0; q < 8; ++q) { mu += red1[row][q]; m2 += red2[row][q]; }
      mu *= (1.f / 64.f); m2 *= (1.f / 64.f);
      float stdv = sqrtf(m2 - mu * mu + EPS_);
      float xh[8], gy[8];
      float sgy = 0.f, sgx = 0.f;
      for (int q = 0; q < 8; ++q) {
        int c = sg * 8 + q;
        xh[q] = (z[q] - mu) / stdv;
        gy[q] = (gW[c] * xh[q] + gB[c] - tg[q]) * gW[c];
        sgy += gy[q]; sgx += gy[q] * xh[q];
      }
      __syncthreads();
      red1[row][sg] = sgy; red2[row][sg] = sgx;
      __syncthreads();
      float Sgy = 0.f, Sgx = 0.f;
      for (int q = 0; q < 8; ++q) { Sgy += red1[row][q]; Sgx += red2[row][q]; }
      float inv = 1.0f / (64.0f * stdv);
      for (int q = 0; q < 8; ++q) {
        int c = sg * 8 + q;
        grads[row][c] = (64.0f * gy[q] - Sgy - xh[q] * Sgx) * inv;
      }
    }
    __syncthreads();

    // ---- M = -(e .* tril(Attn) + tril(e)), Attn = xq @ xk^T (wave 0) ----
    if (wv == 0) {
      v8f a8; for (int r = 0; r < 8; ++r) a8[r] = 0.f;
      for (int ks = 0; ks < 16; ++ks) {
        int k = ks * 4 + kh * 2;
        v2f av, bv;
        av.x = xqs[ln16][k]; av.y = xqs[ln16][k + 1];   // A[m][k], m=ln16
        bv.x = xks[ln16][k]; bv.y = xks[ln16][k + 1];   // B[k][n] = xk[n][k], n=ln16
        a8 = wmma_f32k4(av, bv, a8);
      }
      for (int r = 0; r < 8; ++r) {
        int m = r + (kh << 3);
        float e = tokv[m] * lrv[ln16] * (1.0f / 64.0f);
        Ms[m][ln16] = (m >= ln16) ? (-e * (a8[r] + 1.0f)) : 0.0f;
      }
    }
    __syncthreads();

    // ---- Z1_bar = xq@W1 + M@grad + b1 (overwrite Z1s) ----
    {
      v8f a8; for (int r = 0; r < 8; ++r) a8[r] = 0.f;
      int c = wv * 16 + ln16;
      for (int ks = 0; ks < 16; ++ks) {
        int k = ks * 4 + kh * 2;
        v2f av, bv;
        av.x = xqs[ln16][k]; av.y = xqs[ln16][k + 1];
        bv.x = W1s[k][c];    bv.y = W1s[k + 1][c];
        a8 = wmma_f32k4(av, bv, a8);
      }
      for (int ks = 0; ks < 4; ++ks) {
        int k = ks * 4 + kh * 2;
        v2f av, bv;
        av.x = Ms[ln16][k];  av.y = Ms[ln16][k + 1];
        bv.x = grads[k][c];  bv.y = grads[k + 1][c];
        a8 = wmma_f32k4(av, bv, a8);
      }
      for (int r = 0; r < 8; ++r) { int m = r + (kh << 3); Z1s[m][c] = a8[r] + b1s[c]; }
    }
    __syncthreads();

    // ---- out = xq + ln_fwd(Z1_bar) ----
    {
      int row = tid >> 3, sg = tid & 7;
      float z[8]; float s1 = 0.f, s2 = 0.f;
      for (int q = 0; q < 8; ++q) { int c = sg * 8 + q; float t = Z1s[row][c]; z[q] = t; s1 += t; s2 += t * t; }
      red1[row][sg] = s1; red2[row][sg] = s2;
      __syncthreads();
      float mu = 0.f, m2 = 0.f;
      for (int q = 0; q < 8; ++q) { mu += red1[row][q]; m2 += red2[row][q]; }
      mu *= (1.f / 64.f); m2 *= (1.f / 64.f);
      float stdv = sqrtf(m2 - mu * mu + EPS_);
      size_t obase = ((size_t)(b * L_ + n * 16 + row)) * W_ + h * 64;
      for (int q = 0; q < 8; ++q) {
        int c = sg * 8 + q;
        outp[obase + c] = xqs[row][c] + gW[c] * ((z[q] - mu) / stdv) + gB[c];
      }
    }

    // ---- W1 -= (last_eta*xk)^T @ grad ; wave wv owns rows [16wv,16wv+16) ----
    {
      int d0 = wv * 16;
      int drow = d0 + ln16;
      for (int nt = 0; nt < 4; ++nt) {
        int ncol = nt * 16 + ln16;
        v8f a8;
        for (int r = 0; r < 8; ++r) a8[r] = W1s[d0 + r + (kh << 3)][ncol];
        for (int ks = 0; ks < 4; ++ks) {
          int k = ks * 4 + kh * 2;
          v2f av, bv;
          av.x = -lev[k]     * xks[k][drow];       // A[d][k] = -le[k]*xk[k][d]
          av.y = -lev[k + 1] * xks[k + 1][drow];
          bv.x = grads[k][ncol]; bv.y = grads[k + 1][ncol];
          a8 = wmma_f32k4(av, bv, a8);
        }
        for (int r = 0; r < 8; ++r) W1s[d0 + r + (kh << 3)][ncol] = a8[r];
      }
    }
    if (tid < 64) {
      float s = 0.f;
      for (int j = 0; j < 16; ++j) s += lev[j] * grads[j][tid];
      b1s[tid] -= s;
    }
    __syncthreads();
  }
}

// =====================================================================
// In-place row LayerNorm with affine (post_norm)
// =====================================================================
__global__ __launch_bounds__(256) void postnorm_k(
    float* __restrict__ x, const float* __restrict__ pw, const float* __restrict__ pb)
{
  __shared__ float rA[256], rB[256];
  int row = blockIdx.x, tid = threadIdx.x;
  size_t base = (size_t)row * W_;
  float v[8];
  float s1 = 0.f, s2 = 0.f;
  for (int q = 0; q < 8; ++q) {
    float t = x[base + tid + 256 * q];
    v[q] = t; s1 += t; s2 += t * t;
  }
  rA[tid] = s1; rB[tid] = s2;
  __syncthreads();
  for (int off = 128; off > 0; off >>= 1) {
    if (tid < off) { rA[tid] += rA[tid + off]; rB[tid] += rB[tid + off]; }
    __syncthreads();
  }
  float mu  = rA[0] * (1.0f / 2048.0f);
  float var = rB[0] * (1.0f / 2048.0f) - mu * mu;
  float inv = rsqrtf(var + EPS_);
  for (int q = 0; q < 8; ++q) {
    int c = tid + 256 * q;
    x[base + c] = (v[q] - mu) * inv * pw[c] + pb[c];
  }
}

// =====================================================================
extern "C" void kernel_launch(void* const* d_in, const int* in_sizes, int n_in,
                              void* d_out, int out_size, void* d_ws, size_t ws_size,
                              hipStream_t stream)
{
  const float* hs  = (const float*)d_in[0];
  const float* q_w = (const float*)d_in[1];
  const float* v_w = (const float*)d_in[2];
  const float* o_w = (const float*)d_in[3];
  const float* cqw = (const float*)d_in[4];
  const float* cqb = (const float*)d_in[5];
  const float* ckw = (const float*)d_in[6];
  const float* ckb = (const float*)d_in[7];
  const float* lrw = (const float*)d_in[8];
  const float* lrb = (const float*)d_in[9];
  const float* lti = (const float*)d_in[10];
  const float* nw  = (const float*)d_in[11];
  const float* nb  = (const float*)d_in[12];
  const float* W1i = (const float*)d_in[13];
  const float* b1i = (const float*)d_in[14];
  const float* pw  = (const float*)d_in[15];
  const float* pb  = (const float*)d_in[16];
  float* outp = (float*)d_out;

  const size_t NE = (size_t)B_ * L_ * W_;   // 16,777,216
  float* ws = (float*)d_ws;
  float* xq_lin   = ws;                 // NE
  float* XQh      = ws + NE;            // NE
  float* XKh      = XQh + NE;           // NE
  float* XVh      = XKh + NE;           // NE
  float* lrp      = XVh + NE;           // B*NH*L = 262144
  float* scan_out = xq_lin;             // alias: xq_lin dead after conv_rope
  if (ws_size < (4 * NE + (size_t)B_ * NH_ * L_) * sizeof(float)) return;

  const int M = B_ * L_;                // 8192
  dim3 ggrid(W_ / 128, M / 128);        // (16, 64)
  const size_t gemm_lds = (size_t)(4 * STG_TILE) * sizeof(float);  // 67.6 KB

  // 1) xq = hs @ q_w^T (row-major);  XV = hs @ v_w^T (scatter to head/chunk)
  gemm_bf16split<<<ggrid, 256, gemm_lds, stream>>>(hs, q_w, xq_lin, M, W_, W_, 0);
  gemm_bf16split<<<ggrid, 256, gemm_lds, stream>>>(hs, v_w, XVh,    M, W_, W_, 1);

  // 2) causal dwconv + RoPE -> XQh, XKh
  conv_rope<<<(B_ * L_ * W_ / 2) / 256, 256, 0, stream>>>(xq_lin, cqw, cqb, ckw, ckb, XQh, XKh);

  // 3) ttt_lr gates
  ttt_lr_k<<<B_ * L_, 256, 0, stream>>>(hs, lrw, lrb, lrp);

  // 4) sequential TTT scan (one block per (b,h))
  ttt_scan<<<B_ * NH_, 128, 0, stream>>>(XQh, XKh, XVh, lrp, lti, nw, nb, W1i, b1i, scan_out);

  // 5) post layernorm (in place)
  postnorm_k<<<B_ * L_, 256, 0, stream>>>(scan_out, pw, pb);

  // 6) final projection -> d_out
  gemm_bf16split<<<ggrid, 256, gemm_lds, stream>>>(scan_out, o_w, outp, M, W_, W_, 0);
}
